// AccessibilitySVIGNN_18597208392406
// MI455X (gfx1250) — matrix operand order repaired and checked by reading, not compiled, not run on hardware
//
#include <hip/hip_runtime.h>
#include <hip/hip_bf16.h>
#include <math.h>

// ---------------- constants ----------------
#define NN 100000
#define EE 1600000
#define ENT (EE + NN)        // edges + self loops
#define DIN 256
#define HID 64

typedef __attribute__((ext_vector_type(16))) _Float16 v16h;
typedef __attribute__((ext_vector_type(8)))  float    v8f;
typedef __attribute__((ext_vector_type(2)))  _Float16 h2v;

#define TPB 256

// ---------------- helpers ----------------
__device__ __forceinline__ unsigned f2ord(float f) {
    unsigned u = __float_as_uint(f);
    return (u & 0x80000000u) ? ~u : (u | 0x80000000u);
}
__device__ __forceinline__ float ord2f(unsigned u) {
    return (u & 0x80000000u) ? __uint_as_float(u & 0x7FFFFFFFu) : __uint_as_float(~u);
}
#define ORD_NEG_INF 0x007FFFFFu   // f2ord(-inf)

__device__ __forceinline__ float lrelu(float x) { return x > 0.0f ? x : 0.2f * x; }

// ---------------- conversion kernels ----------------
__global__ void k_f32_to_f16(const float* __restrict__ s, _Float16* __restrict__ d, long n) {
    long t = (long)blockIdx.x * blockDim.x + threadIdx.x;
    if (t < n) d[t] = (_Float16)s[t];
}

// W [K x Co] f32 row-major  ->  Wt [Co x K] f16
__global__ void k_w_transpose_f16(const float* __restrict__ W, _Float16* __restrict__ Wt,
                                  int K, int Co) {
    int t = blockIdx.x * blockDim.x + threadIdx.x;
    if (t >= K * Co) return;
    int k = t / Co, c = t % Co;
    Wt[(long)c * K + k] = (_Float16)W[t];
}

__global__ void k_fill_u32(unsigned* p, unsigned v, long n) {
    long t = (long)blockIdx.x * blockDim.x + threadIdx.x;
    if (t < n) p[t] = v;
}
__global__ void k_fill_f32(float* p, float v, long n) {
    long t = (long)blockIdx.x * blockDim.x + threadIdx.x;
    if (t < n) p[t] = v;
}

// ---------------- WMMA GEMM:  C[M x Co] = A[M x K] * B[K x Co] ----------------
// A is f16 row-major [M x K];  Bt is f16 [Co x K] (i.e. B transposed);  C is f32.
// One wave computes a 16-row x (TN*16)-col strip: A fragment loaded once per
// k-step and reused for TN back-to-back WMMAs (independent accumulators).
// Requires K % 32 == 0, M % 16 == 0, Co % (TN*16) == 0.
template <int TN>
__global__ void k_gemm_wmma(const _Float16* __restrict__ A,
                            const _Float16* __restrict__ Bt,
                            float* __restrict__ C,
                            int M, int K, int Co) {
    const int lane = threadIdx.x & 31;
    const int wave = threadIdx.x >> 5;
    const int colGroups = Co / (16 * TN);
    const int totalTiles = (M >> 4) * colGroups;
    const int tile = blockIdx.x * (blockDim.x >> 5) + wave;
    if (tile >= totalTiles) return;          // uniform per wave: EXEC stays all-ones
    const int rowTile = tile / colGroups;
    const int colGrp  = tile % colGroups;

    const int m    = lane & 15;              // row within A half / column within B half
    const int hi   = lane >> 4;              // 0: lanes 0-15, 1: lanes 16-31
    const int khiA = hi * 8;                 // unused; A loads use khiB (K+16)
    const int khiB = hi * 16;                // B: high lanes hold K+16

    const _Float16* arow = A + (long)(rowTile * 16 + m) * K;
    const _Float16* brow[TN];
#pragma unroll
    for (int j = 0; j < TN; ++j)
        brow[j] = Bt + (long)(colGrp * TN * 16 + j * 16 + m) * K + khiB;

    v8f acc[TN];
#pragma unroll
    for (int j = 0; j < TN; ++j) acc[j] = (v8f){};

    for (int k0 = 0; k0 < K; k0 += 32) {
        v16h a;
#pragma unroll
        for (int i = 0; i < 8; ++i) {
            // A, contiguous k order: VGPR i holds K = 2i (+16 hi lanes)
            int kk = 2 * i + khiB;
            h2v pa = *(const h2v*)(arow + k0 + kk);
            a[2 * i] = pa[0]; a[2 * i + 1] = pa[1];
        }
#pragma unroll
        for (int j = 0; j < TN; ++j) {
            v16h b;
#pragma unroll
            for (int i = 0; i < 8; ++i) {
                // B, same contiguous k order as A: lane = N, VGPR i holds K = 2i (+16 hi lanes)
                h2v pb = *(const h2v*)(brow[j] + k0 + 2 * i);
                b[2 * i] = pb[0]; b[2 * i + 1] = pb[1];
            }
            acc[j] = __builtin_amdgcn_wmma_f32_16x16x32_f16(false, a, false, b,
                                                            (short)0, acc[j], false, false);
        }
    }

    // C/D layout: lanes 0-15 -> N=lane, M=r ; lanes 16-31 -> N=lane-16, M=8+r
    const int mrow0 = rowTile * 16 + hi * 8;
#pragma unroll
    for (int j = 0; j < TN; ++j) {
        const int col = colGrp * TN * 16 + j * 16 + m;
#pragma unroll
        for (int r = 0; r < 8; ++r)
            C[(long)(mrow0 + r) * Co + col] = acc[j][r];
    }
}

// ---------------- GAT pieces ----------------
// a_s[n,h] = sum_d hf[n,h,d]*att_s[h,d]  (hf is [N, heads*64])
__global__ void k_att_dots(const float* __restrict__ hf,
                           const float* __restrict__ atts, const float* __restrict__ attd,
                           float* __restrict__ as_, float* __restrict__ ad_, int heads) {
    int t = blockIdx.x * blockDim.x + threadIdx.x;
    if (t >= NN * heads) return;
    int n = t / heads, h = t % heads;
    const float* hp = hf + ((long)n * heads + h) * HID;
    const float* sp = atts + h * HID;
    const float* dp = attd + h * HID;
    float s = 0.f, d = 0.f;
#pragma unroll 4
    for (int i = 0; i < HID; i += 4) {
        float4 v = *(const float4*)(hp + i);
        float4 a = *(const float4*)(sp + i);
        float4 b = *(const float4*)(dp + i);
        s += v.x * a.x + v.y * a.y + v.z * a.z + v.w * a.w;
        d += v.x * b.x + v.y * b.y + v.z * b.z + v.w * b.w;
    }
    as_[t] = s; ad_[t] = d;
}

__global__ void k_edge_max(const long long* __restrict__ ei,
                           const float* __restrict__ as_, const float* __restrict__ ad_,
                           unsigned* __restrict__ mx, int heads) {
    int t = blockIdx.x * blockDim.x + threadIdx.x;
    if (t >= ENT) return;
    int s, d;
    if (t < EE) { s = (int)ei[t]; d = (int)ei[EE + t]; } else { s = d = t - EE; }
    for (int h = 0; h < heads; ++h) {
        float e = lrelu(as_[(long)s * heads + h] + ad_[(long)d * heads + h]);
        atomicMax(&mx[(long)d * heads + h], f2ord(e));
    }
}

__global__ void k_edge_expsum(const long long* __restrict__ ei,
                              const float* __restrict__ as_, const float* __restrict__ ad_,
                              const unsigned* __restrict__ mx,
                              float* __restrict__ alpha, float* __restrict__ sum, int heads) {
    int t = blockIdx.x * blockDim.x + threadIdx.x;
    if (t >= ENT) return;
    int s, d;
    if (t < EE) { s = (int)ei[t]; d = (int)ei[EE + t]; } else { s = d = t - EE; }
    for (int h = 0; h < heads; ++h) {
        float e = lrelu(as_[(long)s * heads + h] + ad_[(long)d * heads + h]);
        float m = ord2f(mx[(long)d * heads + h]);
        float ex = expf(e - m);
        alpha[(long)t * heads + h] = ex;
        atomicAdd(&sum[(long)d * heads + h], ex);
    }
}

// acc[dst,h,d] += hf[src,h,d] * alpha/(sum+eps)   (thread per (edge,head))
__global__ void k_edge_msg(const long long* __restrict__ ei,
                           const float* __restrict__ hf, const float* __restrict__ alpha,
                           const float* __restrict__ sum, float* __restrict__ acc, int heads) {
    long t = (long)blockIdx.x * blockDim.x + threadIdx.x;
    if (t >= (long)ENT * heads) return;
    int h = (int)(t % heads);
    int e = (int)(t / heads);
    int s, d;
    if (e < EE) { s = (int)ei[e]; d = (int)ei[EE + e]; } else { s = d = e - EE; }
    float w = alpha[(long)e * heads + h] / (sum[(long)d * heads + h] + 1e-16f);
    const float* hs = hf + ((long)s * heads + h) * HID;
    float* ap = acc + ((long)d * heads + h) * HID;
#pragma unroll
    for (int i = 0; i < HID; i += 4) {
        float4 v = *(const float4*)(hs + i);
        atomicAdd(ap + i + 0, v.x * w);
        atomicAdd(ap + i + 1, v.y * w);
        atomicAdd(ap + i + 2, v.z * w);
        atomicAdd(ap + i + 3, v.w * w);
    }
}

// h_out[n,d] = relu(mean_h acc[n,h,d] + bias[d]) -> written as f16 (next layer's GEMM input)
__global__ void k_gat_final(const float* __restrict__ acc, const float* __restrict__ bias,
                            _Float16* __restrict__ outh, int heads) {
    int t = blockIdx.x * blockDim.x + threadIdx.x;
    if (t >= NN * HID) return;
    int n = t >> 6, d = t & 63;
    float s = 0.f;
    for (int h = 0; h < heads; ++h) s += acc[((long)n * heads + h) * HID + d];
    s = s * (1.0f / heads) + bias[d];
    outh[t] = (_Float16)(s > 0.f ? s : 0.f);
}

// ---------------- GCN pieces ----------------
__global__ void k_deg_edges(const long long* __restrict__ ei, float* __restrict__ deg) {
    int e = blockIdx.x * blockDim.x + threadIdx.x;
    if (e >= EE) return;
    atomicAdd(&deg[(int)ei[EE + e]], 1.0f);
}
__global__ void k_dinv(float* __restrict__ deg) {
    int n = blockIdx.x * blockDim.x + threadIdx.x;
    if (n < NN) deg[n] = rsqrtf(deg[n]);   // deg >= 1 always (self loop)
}
__global__ void k_gcn_scatter(const long long* __restrict__ ei,
                              const float* __restrict__ hf, const float* __restrict__ dinv,
                              float* __restrict__ acc) {
    int e = blockIdx.x * blockDim.x + threadIdx.x;
    if (e >= EE) return;
    int s = (int)ei[e], d = (int)ei[EE + e];
    float nrm = dinv[s] * dinv[d];
    const float* hs = hf + (long)s * 32;
    float* ap = acc + (long)d * 32;
#pragma unroll
    for (int i = 0; i < 32; i += 4) {
        float4 v = *(const float4*)(hs + i);
        atomicAdd(ap + i + 0, v.x * nrm);
        atomicAdd(ap + i + 1, v.y * nrm);
        atomicAdd(ap + i + 2, v.z * nrm);
        atomicAdd(ap + i + 3, v.w * nrm);
    }
}
// h3 = relu(acc + self-loop term + bg)
__global__ void k_gcn_final(const float* __restrict__ acc, const float* __restrict__ hf,
                            const float* __restrict__ dinv, const float* __restrict__ bg,
                            float* __restrict__ h3) {
    int t = blockIdx.x * blockDim.x + threadIdx.x;
    if (t >= NN * 32) return;
    int n = t >> 5, d = t & 31;
    float di = dinv[n];
    float s = acc[t] + hf[t] * di * di + bg[d];
    h3[t] = s > 0.f ? s : 0.f;
}

// ---------------- MLP head ----------------
__global__ void k_mlp(const float* __restrict__ h3,
                      const float* __restrict__ Wh1, const float* __restrict__ bh1,
                      const float* __restrict__ Wh2, const float* __restrict__ bh2,
                      float* __restrict__ out) {
    int n = blockIdx.x * blockDim.x + threadIdx.x;
    if (n >= NN) return;
    const float* x = h3 + (long)n * 32;
    float xr[32];
#pragma unroll
    for (int i = 0; i < 32; i += 4) { float4 v = *(const float4*)(x + i);
        xr[i] = v.x; xr[i + 1] = v.y; xr[i + 2] = v.z; xr[i + 3] = v.w; }
    float o = bh2[0];
#pragma unroll
    for (int j = 0; j < 16; ++j) {
        float z = bh1[j];
#pragma unroll
        for (int i = 0; i < 32; ++i) z += xr[i] * Wh1[i * 16 + j];
        z = z > 0.f ? z : 0.f;
        o += z * Wh2[j];
    }
    out[n] = 1.0f / (1.0f + expf(-o));
}

// ---------------- host side ----------------
static inline int gblk(long n) { return (int)((n + TPB - 1) / TPB); }
static inline size_t alup(size_t x) { return (x + 255) & ~(size_t)255; }

extern "C" void kernel_launch(void* const* d_in, const int* in_sizes, int n_in,
                              void* d_out, int out_size, void* d_ws, size_t ws_size,
                              hipStream_t stream) {
    const float*     x   = (const float*)d_in[0];
    const long long* ei  = (const long long*)d_in[1];
    const float* W1  = (const float*)d_in[2];
    const float* a1s = (const float*)d_in[3];
    const float* a1d = (const float*)d_in[4];
    const float* b1  = (const float*)d_in[5];
    const float* W2  = (const float*)d_in[6];
    const float* a2s = (const float*)d_in[7];
    const float* a2d = (const float*)d_in[8];
    const float* b2  = (const float*)d_in[9];
    const float* Wg  = (const float*)d_in[10];
    const float* bg  = (const float*)d_in[11];
    const float* Wh1 = (const float*)d_in[12];
    const float* bh1 = (const float*)d_in[13];
    const float* Wh2 = (const float*)d_in[14];
    const float* bh2 = (const float*)d_in[15];
    float* out = (float*)d_out;

    // workspace carve-up
    char* w = (char*)d_ws;
    size_t off = 0;
    _Float16* xh  = (_Float16*)(w + off); off += alup((size_t)NN * DIN * 2);   // f16 layer input
    _Float16* wt  = (_Float16*)(w + off); off += alup((size_t)DIN * DIN * 2);  // f16 W^T
    float*    hf  = (float*)(w + off);    off += alup((size_t)NN * DIN * 4);   // GEMM out [N, heads*64]
    float*    as_ = (float*)(w + off);    off += alup((size_t)NN * 4 * 4);
    float*    ad_ = (float*)(w + off);    off += alup((size_t)NN * 4 * 4);
    unsigned* mx  = (unsigned*)(w + off); off += alup((size_t)NN * 4 * 4);
    float*    sum = (float*)(w + off);    off += alup((size_t)NN * 4 * 4);
    float*    alp = (float*)(w + off);    off += alup((size_t)ENT * 4 * 4);
    float*    acc = (float*)(w + off);    off += alup((size_t)NN * DIN * 4);
    float*    h3  = (float*)(w + off);    off += alup((size_t)NN * 32 * 4);
    float*    deg = (float*)(w + off);    off += alup((size_t)NN * 4);
    (void)ws_size; (void)n_in; (void)in_sizes; (void)out_size;

    const int wavesPerBlk = TPB / 32;

    // ===== GAT layer 1 (heads=4, K=256, Co=256) =====
    k_f32_to_f16<<<gblk((long)NN * DIN), TPB, 0, stream>>>(x, xh, (long)NN * DIN);
    k_w_transpose_f16<<<gblk(DIN * DIN), TPB, 0, stream>>>(W1, wt, DIN, DIN);
    {
        int tiles = (NN / 16) * (DIN / 64);          // TN=4
        k_gemm_wmma<4><<<(tiles + wavesPerBlk - 1) / wavesPerBlk, TPB, 0, stream>>>(
            xh, wt, hf, NN, DIN, DIN);
    }
    k_att_dots<<<gblk((long)NN * 4), TPB, 0, stream>>>(hf, a1s, a1d, as_, ad_, 4);
    k_fill_u32<<<gblk((long)NN * 4), TPB, 0, stream>>>(mx, ORD_NEG_INF, (long)NN * 4);
    k_fill_f32<<<gblk((long)NN * 4), TPB, 0, stream>>>(sum, 0.f, (long)NN * 4);
    k_fill_f32<<<gblk((long)NN * 4 * HID), TPB, 0, stream>>>(acc, 0.f, (long)NN * 4 * HID);
    k_edge_max<<<gblk(ENT), TPB, 0, stream>>>(ei, as_, ad_, mx, 4);
    k_edge_expsum<<<gblk(ENT), TPB, 0, stream>>>(ei, as_, ad_, mx, alp, sum, 4);
    k_edge_msg<<<gblk((long)ENT * 4), TPB, 0, stream>>>(ei, hf, alp, sum, acc, 4);
    k_gat_final<<<gblk((long)NN * HID), TPB, 0, stream>>>(acc, b1, xh, 4);  // h1 (f16) -> xh

    // ===== GAT layer 2 (heads=2, K=64, Co=128) =====
    k_w_transpose_f16<<<gblk(HID * 128), TPB, 0, stream>>>(W2, wt, HID, 128);
    {
        int tiles = (NN / 16) * (128 / 64);          // TN=4
        k_gemm_wmma<4><<<(tiles + wavesPerBlk - 1) / wavesPerBlk, TPB, 0, stream>>>(
            xh, wt, hf, NN, HID, 128);
    }
    k_att_dots<<<gblk((long)NN * 2), TPB, 0, stream>>>(hf, a2s, a2d, as_, ad_, 2);
    k_fill_u32<<<gblk((long)NN * 2), TPB, 0, stream>>>(mx, ORD_NEG_INF, (long)NN * 2);
    k_fill_f32<<<gblk((long)NN * 2), TPB, 0, stream>>>(sum, 0.f, (long)NN * 2);
    k_fill_f32<<<gblk((long)NN * 2 * HID), TPB, 0, stream>>>(acc, 0.f, (long)NN * 2 * HID);
    k_edge_max<<<gblk(ENT), TPB, 0, stream>>>(ei, as_, ad_, mx, 2);
    k_edge_expsum<<<gblk(ENT), TPB, 0, stream>>>(ei, as_, ad_, mx, alp, sum, 2);
    k_edge_msg<<<gblk((long)ENT * 2), TPB, 0, stream>>>(ei, hf, alp, sum, acc, 2);
    k_gat_final<<<gblk((long)NN * HID), TPB, 0, stream>>>(acc, b2, xh, 2);  // h2 (f16) -> xh

    // ===== GCN (K=64, Co=32) =====
    k_w_transpose_f16<<<gblk(HID * 32), TPB, 0, stream>>>(Wg, wt, HID, 32);
    {
        int tiles = (NN / 16) * (32 / 32);           // TN=2
        k_gemm_wmma<2><<<(tiles + wavesPerBlk - 1) / wavesPerBlk, TPB, 0, stream>>>(
            xh, wt, hf, NN, HID, 32);
    }
    k_fill_f32<<<gblk(NN), TPB, 0, stream>>>(deg, 1.0f, NN);               // self loop
    k_deg_edges<<<gblk(EE), TPB, 0, stream>>>(ei, deg);
    k_dinv<<<gblk(NN), TPB, 0, stream>>>(deg);
    k_fill_f32<<<gblk((long)NN * 32), TPB, 0, stream>>>(acc, 0.f, (long)NN * 32);
    k_gcn_scatter<<<gblk(EE), TPB, 0, stream>>>(ei, hf, deg, acc);
    k_gcn_final<<<gblk((long)NN * 32), TPB, 0, stream>>>(acc, hf, deg, bg, h3);

    // ===== MLP head =====
    k_mlp<<<gblk(NN), TPB, 0, stream>>>(h3, Wh1, bh1, Wh2, bh2, out);
}
